// EncoderLayer_76072460747307
// MI455X (gfx1250) — matrix operand order repaired and checked by reading, not compiled, not run on hardware
//
#include <hip/hip_runtime.h>

// ---------------------------------------------------------------------------
// Types & helpers
// ---------------------------------------------------------------------------
typedef __bf16 bf16x16 __attribute__((ext_vector_type(16)));
typedef float  floatx8 __attribute__((ext_vector_type(8)));

__device__ __forceinline__ unsigned short f2bf_bits(float f) {
  unsigned u = __builtin_bit_cast(unsigned, f);
  unsigned r = u + 0x7FFFu + ((u >> 16) & 1u);  // round-to-nearest-even
  return (unsigned short)(r >> 16);
}
__device__ __forceinline__ __bf16 f2bf(float f) {
  return __builtin_bit_cast(__bf16, f2bf_bits(f));
}

__device__ __forceinline__ floatx8 wmma_bf16(bf16x16 a, bf16x16 b, floatx8 c) {
  // D(f32,16x16) = A(bf16,16x32) * B(bf16,32x16) + C
  return __builtin_amdgcn_wmma_f32_16x16x32_bf16(false, a, false, b,
                                                 (short)0, c, false, false);
}

union FragU {
  bf16x16 v;
  uint4   q[2];
  __bf16  h[16];
};

// --- async global->LDS staging (gfx1250 GLOBAL_LOAD_ASYNC_TO_LDS_B128),
// --- guarded so the file compiles on toolchains without the builtins.
#if __has_builtin(__builtin_amdgcn_global_load_async_to_lds_b128) && \
    __has_builtin(__builtin_amdgcn_s_wait_asynccnt)
#define HAS_ASYNC_LDS 1
typedef int v4i_ __attribute__((vector_size(4 * sizeof(int))));
typedef __attribute__((address_space(1))) v4i_* g4p;  // global int4*
typedef __attribute__((address_space(3))) v4i_* l4p;  // LDS int4*
#else
#define HAS_ASYNC_LDS 0
#endif

__device__ __forceinline__ void cp16(__bf16* lds, const __bf16* glob) {
#if HAS_ASYNC_LDS
  __builtin_amdgcn_global_load_async_to_lds_b128((g4p)glob, (l4p)lds, 0, 0);
#else
  *(uint4*)lds = *(const uint4*)glob;
#endif
}
__device__ __forceinline__ void staging_fence() {
#if HAS_ASYNC_LDS
  __builtin_amdgcn_s_wait_asynccnt(0);
#endif
  __syncthreads();
}

static constexpr int D_MODEL = 768;
static constexpr int HID     = 3072;
static constexpr int N_HEADS = 12;
static constexpr int DK      = 64;

// ---------------------------------------------------------------------------
// f32 -> bf16 convert (vectorized x4)
// ---------------------------------------------------------------------------
__global__ void cvt_kernel(const float* __restrict__ in,
                           __bf16* __restrict__ out, int n4) {
  int i = blockIdx.x * blockDim.x + threadIdx.x;
  if (i >= n4) return;
  float4 f = reinterpret_cast<const float4*>(in)[i];
  unsigned lo = (unsigned)f2bf_bits(f.x) | ((unsigned)f2bf_bits(f.y) << 16);
  unsigned hi = (unsigned)f2bf_bits(f.z) | ((unsigned)f2bf_bits(f.w) << 16);
  reinterpret_cast<uint2*>(out)[i] = make_uint2(lo, hi);
}

// ---------------------------------------------------------------------------
// Tiled transpose + convert: Out_bf16[N][K] = In_f32[K][N]^T
// 32x32 tiles through LDS, both sides coalesced. Block = 256 threads.
// ---------------------------------------------------------------------------
__global__ void cvt_t_kernel(const float* __restrict__ in,
                             __bf16* __restrict__ out, int K, int N) {
  __shared__ float tile[32][33];
  const int tx = threadIdx.x & 31, ty = threadIdx.x >> 5;
  const int n0 = blockIdx.x * 32, k0 = blockIdx.y * 32;
#pragma unroll
  for (int i = 0; i < 4; ++i) {
    int kk = ty + i * 8;
    tile[kk][tx] = in[(size_t)(k0 + kk) * N + n0 + tx];
  }
  __syncthreads();
#pragma unroll
  for (int i = 0; i < 4; ++i) {
    int nn = ty + i * 8;
    out[(size_t)(n0 + nn) * K + k0 + tx] = f2bf(tile[tx][nn]);
  }
}

// ---------------------------------------------------------------------------
// Tiled WMMA GEMM:  C[M,N] = A_bf16[M,K] @ Bt_bf16[N,K]^T + bias  (opt ReLU)
// B is PRE-TRANSPOSED (weights), so B fragments are two contiguous
// global_load_b128 per WMMA operand — no LDS, no shuffles.
// Block tile: BM=128 x BN=64 x BK=32, 256 threads = 8 waves,
// each wave a 2x2 grid of 16x16 tiles (4 WMMA / wave / K-chunk).
// OUT_T writes C transposed (used to produce V^T for attention).
// ---------------------------------------------------------------------------
template <bool OUT_BF16, bool RELU, bool OUT_T>
__global__ void gemm_wmma(const __bf16* __restrict__ A,
                          const __bf16* __restrict__ Bt,
                          const float* __restrict__ bias,
                          void* __restrict__ Cout,
                          int M, int N, int K) {
  __shared__ alignas(16) __bf16 As[128][32];  // [m][k]

  const int tid  = threadIdx.x;
  const int wave = tid >> 5;
  const int lane = tid & 31;
  const int lr   = lane & 15;
  const int lh   = lane >> 4;
  const int tm   = (wave & 3) * 32;   // wave's m origin within tile
  const int tn   = (wave >> 2) * 32;  // wave's n origin within tile

  const int m0 = blockIdx.y * 128;
  const int n0 = blockIdx.x * 64;

  floatx8 acc[2][2] = {};

  for (int k0 = 0; k0 < K; k0 += 32) {
    // ---- stage A tile: 128x32 bf16, 16 elems (2 x b128) per thread ----
    {
      int idx = tid * 16;
      int r = idx >> 5, c = idx & 31;
      const __bf16* src = &A[(size_t)(m0 + r) * K + k0 + c];
      cp16(&As[r][c], src);
      cp16(&As[r][c + 8], src + 8);
      if (k0 + 32 < K) __builtin_prefetch(src + 32, 0, 1);
    }
    staging_fence();

    // A fragments: row = tm + im*16 + lr; K halves at lh*8 and 16+lh*8
    FragU a[2];
#pragma unroll
    for (int im = 0; im < 2; ++im) {
      const __bf16* p = &As[tm + im * 16 + lr][lh * 8];
      a[im].q[0] = *(const uint4*)p;
      a[im].q[1] = *(const uint4*)(p + 16);
    }
    // B fragments: col = tn + in*16 + lr; K halves at lh*8 and 16+lh*8
    // -> two 8-bf16 runs straight from global Bt[col][k0 + lh*8 ...]
    FragU b[2];
#pragma unroll
    for (int in = 0; in < 2; ++in) {
      const __bf16* p =
          &Bt[(size_t)(n0 + tn + in * 16 + lr) * K + k0 + lh * 8];
      b[in].q[0] = *(const uint4*)p;
      b[in].q[1] = *(const uint4*)(p + 16);
      if (k0 + 32 < K) __builtin_prefetch(p + 32, 0, 1);
    }
#pragma unroll
    for (int im = 0; im < 2; ++im)
#pragma unroll
      for (int in = 0; in < 2; ++in)
        acc[im][in] = wmma_bf16(a[im].v, b[in].v, acc[im][in]);
    __syncthreads();
  }

  // Epilogue: C layout -> col = lr, row = j + 8*lh
#pragma unroll
  for (int in = 0; in < 2; ++in) {
    const int col = n0 + tn + in * 16 + lr;
    const float bv = bias[col];
#pragma unroll
    for (int im = 0; im < 2; ++im) {
#pragma unroll
      for (int j = 0; j < 8; ++j) {
        int row = m0 + tm + im * 16 + j + lh * 8;
        float v = acc[im][in][j] + bv;
        if (RELU) v = fmaxf(v, 0.0f);
        size_t idx = OUT_T ? (size_t)col * M + row : (size_t)row * N + col;
        if (OUT_BF16)
          ((__bf16*)Cout)[idx] = f2bf(v);
        else
          ((float*)Cout)[idx] = v;
      }
    }
  }
}

// ---------------------------------------------------------------------------
// Flash attention (bf16 WMMA, f32 accumulate, online softmax).
// Block = 256 threads = 8 waves; each wave owns a 16-row Q tile (128/block).
// 32 keys per iteration. K staged row-major [key][dk]; V consumed from the
// pre-transposed Vt[D][M], staged [dk][key] — all fragments are contiguous
// b128 LDS reads; the only scalar LDS traffic left is the inherent P
// relayout of flash attention.
// ---------------------------------------------------------------------------
__global__ void attn_kernel(const __bf16* __restrict__ Q,
                            const __bf16* __restrict__ Kt,
                            const __bf16* __restrict__ Vt,
                            const int* __restrict__ mask,
                            __bf16* __restrict__ O, int T, int ldv) {
  __shared__ alignas(16) __bf16 Ks[32][64];     // [key][dk]
  __shared__ alignas(16) __bf16 Vts[64][32];    // [dk][key]
  __shared__ alignas(16) __bf16 Ps[8][16][32];  // per-wave P tile

  const int tid  = threadIdx.x;
  const int wave = tid >> 5;
  const int lane = tid & 31;
  const int lr   = lane & 15;
  const int lh   = lane >> 4;

  const int h  = blockIdx.y;
  const int bz = blockIdx.z;
  const int q0 = blockIdx.x * 128 + wave * 16;
  const size_t baseRow = (size_t)bz * T;
  const int D = D_MODEL;

  // Q fragments (16 x 64), kept in registers for the whole K loop.
  FragU qa[2];
#pragma unroll
  for (int d = 0; d < 2; ++d) {
    const __bf16* p = &Q[(baseRow + q0 + lr) * D + h * DK + d * 32 + lh * 8];
    qa[d].q[0] = *(const uint4*)p;
    qa[d].q[1] = *(const uint4*)(p + 16);
  }

  floatx8 o[4] = {};
  float mi[8], li[8];
#pragma unroll
  for (int j = 0; j < 8; ++j) { mi[j] = -__builtin_inff(); li[j] = 0.0f; }

  const float scale = 0.125f;  // 1/sqrt(64)

  for (int kc = 0; kc < T; kc += 32) {
    __syncthreads();  // previous iteration done before restaging
    {
      // K chunk: 32 keys x 64 dk, row-major
      int idx = tid * 8;
      int r = idx >> 6, c = idx & 63;
      cp16(&Ks[r][c], &Kt[(baseRow + kc + r) * D + h * DK + c]);
      // V^T chunk: 64 dk-rows x 32 keys (contiguous along keys in Vt)
      int r2 = idx >> 5, c2 = idx & 31;
      cp16(&Vts[r2][c2], &Vt[(size_t)(h * DK + r2) * ldv + baseRow + kc + c2]);
    }
    staging_fence();

    // ---- S = Q @ K^T for two 16-key subtiles ----
    floatx8 s[2];
#pragma unroll
    for (int kn = 0; kn < 2; ++kn) {
      floatx8 c = {};
#pragma unroll
      for (int d = 0; d < 2; ++d) {
        FragU kb;  // B(kk,key)=K[key][d*32+kk]: contiguous in Ks row
        const __bf16* p = &Ks[kn * 16 + lr][d * 32 + lh * 8];
        kb.q[0] = *(const uint4*)p;
        kb.q[1] = *(const uint4*)(p + 16);
        c = wmma_bf16(qa[d].v, kb.v, c);
      }
      int kt = kc + kn * 16 + lr;
      bool valid = mask[(size_t)bz * T + kt] != 0;
#pragma unroll
      for (int j = 0; j < 8; ++j)
        c[j] = valid ? c[j] * scale : -__builtin_inff();
      s[kn] = c;
    }

    // ---- online softmax; each row lives across one 16-lane half ----
#pragma unroll
    for (int j = 0; j < 8; ++j) {
      float rmax = fmaxf(s[0][j], s[1][j]);
#pragma unroll
      for (int off = 8; off; off >>= 1)
        rmax = fmaxf(rmax, __shfl_xor(rmax, off, 16));
      float mnew = fmaxf(mi[j], rmax);
      float corr, p0, p1;
      if (mnew == -__builtin_inff()) {
        corr = 0.0f; p0 = 0.0f; p1 = 0.0f;
      } else {
        corr = (mi[j] == -__builtin_inff()) ? 0.0f : __expf(mi[j] - mnew);
        p0 = __expf(s[0][j] - mnew);
        p1 = __expf(s[1][j] - mnew);
      }
      float rs = p0 + p1;
#pragma unroll
      for (int off = 8; off; off >>= 1) rs += __shfl_xor(rs, off, 16);
      li[j] = li[j] * corr + rs;
      mi[j] = mnew;
#pragma unroll
      for (int nd = 0; nd < 4; ++nd) o[nd][j] *= corr;
      int m = j + lh * 8;
      Ps[wave][m][lr]      = f2bf(p0);
      Ps[wave][m][16 + lr] = f2bf(p1);
    }
    __syncthreads();  // P visible (C-layout -> A-layout via LDS)

    // ---- O += P(16x32) @ V(32x64) ----
    FragU pa;
    {
      const __bf16* p = &Ps[wave][lr][lh * 8];
      pa.q[0] = *(const uint4*)p;
      pa.q[1] = *(const uint4*)(p + 16);
    }
#pragma unroll
    for (int nd = 0; nd < 4; ++nd) {
      FragU vb;  // B(kk,n)=Vt[nd*16+n][kk]: contiguous in Vts row
      const __bf16* p = &Vts[nd * 16 + lr][lh * 8];
      vb.q[0] = *(const uint4*)p;
      vb.q[1] = *(const uint4*)(p + 16);
      o[nd] = wmma_bf16(pa.v, vb.v, o[nd]);
    }
  }

  // ---- normalize and write out ----
#pragma unroll
  for (int nd = 0; nd < 4; ++nd) {
    int col = h * DK + nd * 16 + lr;
#pragma unroll
    for (int j = 0; j < 8; ++j) {
      int row = q0 + j + lh * 8;
      float dn = li[j];
      float val = (dn > 0.0f) ? o[nd][j] / dn : 0.0f;
      O[(baseRow + row) * D + col] = f2bf(val);
    }
  }
}

// ---------------------------------------------------------------------------
// Residual + LayerNorm over D=768; also emits bf16 copy if outb != nullptr.
// ---------------------------------------------------------------------------
__global__ void ln_kernel(const float* __restrict__ resid,
                          const float* __restrict__ y,
                          const float* __restrict__ g,
                          const float* __restrict__ bb,
                          float* __restrict__ outf,
                          __bf16* __restrict__ outb) {
  const int D = D_MODEL;
  const int row = blockIdx.x, tid = threadIdx.x;
  __shared__ float sred[256];
  const size_t base = (size_t)row * D;

  float v[3], s = 0.0f, ss = 0.0f;
#pragma unroll
  for (int i = 0; i < 3; ++i) {
    int c = tid + i * 256;
    float t = resid[base + c] + y[base + c];
    v[i] = t; s += t; ss += t * t;
  }
  sred[tid] = s; __syncthreads();
  for (int st = 128; st; st >>= 1) {
    if (tid < st) sred[tid] += sred[tid + st];
    __syncthreads();
  }
  float mean = sred[0] * (1.0f / 768.0f);
  __syncthreads();
  sred[tid] = ss; __syncthreads();
  for (int st = 128; st; st >>= 1) {
    if (tid < st) sred[tid] += sred[tid + st];
    __syncthreads();
  }
  float var = sred[0] * (1.0f / 768.0f) - mean * mean;
  float inv = rsqrtf(var + 1e-5f);
#pragma unroll
  for (int i = 0; i < 3; ++i) {
    int c = tid + i * 256;
    float ov = (v[i] - mean) * inv * g[c] + bb[c];
    outf[base + c] = ov;
    if (outb) outb[base + c] = f2bf(ov);
  }
}

// ---------------------------------------------------------------------------
// Launcher
// ---------------------------------------------------------------------------
extern "C" void kernel_launch(void* const* d_in, const int* in_sizes, int n_in,
                              void* d_out, int out_size, void* d_ws,
                              size_t ws_size, hipStream_t stream) {
  const int B = 2, T = 4096, D = D_MODEL, H = HID;
  const int M = B * T;  // 8192 rows

  const float* x    = (const float*)d_in[0];
  const int*   mask = (const int*)d_in[1];
  const float* Wq = (const float*)d_in[2];  const float* bq = (const float*)d_in[3];
  const float* Wk = (const float*)d_in[4];  const float* bk = (const float*)d_in[5];
  const float* Wv = (const float*)d_in[6];  const float* bv = (const float*)d_in[7];
  const float* Wo = (const float*)d_in[8];  const float* bo = (const float*)d_in[9];
  const float* W1 = (const float*)d_in[10]; const float* b1 = (const float*)d_in[11];
  const float* W2 = (const float*)d_in[12]; const float* b2 = (const float*)d_in[13];
  const float* ln1_g = (const float*)d_in[14]; const float* ln1_b = (const float*)d_in[15];
  const float* ln2_g = (const float*)d_in[16]; const float* ln2_b = (const float*)d_in[17];

  // ---- workspace layout (~140 MB; FFN hidden aliases dead Q/K/V/attn) ----
  size_t off = 0;
  auto alloc = [&](size_t bytes) -> void* {
    void* r = (char*)d_ws + off;
    off += (bytes + 255) & ~(size_t)255;
    return r;
  };
  __bf16* xb   = (__bf16*)alloc((size_t)M * D * 2);
  __bf16* Wqt  = (__bf16*)alloc((size_t)D * D * 2);  // [N][K] transposed
  __bf16* Wkt  = (__bf16*)alloc((size_t)D * D * 2);
  __bf16* Wvt  = (__bf16*)alloc((size_t)D * D * 2);
  __bf16* Wot  = (__bf16*)alloc((size_t)D * D * 2);
  __bf16* W1t  = (__bf16*)alloc((size_t)D * H * 2);  // [H][D]
  __bf16* W2t  = (__bf16*)alloc((size_t)H * D * 2);  // [D][H]
  float*  x1   = (float*)alloc((size_t)M * D * 4);
  __bf16* x1b  = (__bf16*)alloc((size_t)M * D * 2);
  float*  tmpf = (float*)alloc((size_t)M * D * 4);   // attn-proj, then ffn2
  __bf16* Qb   = (__bf16*)alloc((size_t)M * D * 2);
  __bf16* Kb   = (__bf16*)alloc((size_t)M * D * 2);
  __bf16* Vtb  = (__bf16*)alloc((size_t)M * D * 2);  // V^T: [D][M]
  __bf16* Ab   = (__bf16*)alloc((size_t)M * D * 2);
  __bf16* hb   = Qb;  // M*H*2 == 4*(M*D*2): reuse Q/K/V/attn region for FFN

  // ---- conversions: x plain; weights transposed+converted ----
  {
    int n4 = M * D / 4;
    cvt_kernel<<<(n4 + 255) / 256, 256, 0, stream>>>(x, xb, n4);
  }
  auto cvt_t = [&](const float* in, __bf16* out, int K, int N) {
    cvt_t_kernel<<<dim3(N / 32, K / 32), 256, 0, stream>>>(in, out, K, N);
  };
  cvt_t(Wq, Wqt, D, D);
  cvt_t(Wk, Wkt, D, D);
  cvt_t(Wv, Wvt, D, D);
  cvt_t(Wo, Wot, D, D);
  cvt_t(W1, W1t, D, H);
  cvt_t(W2, W2t, H, D);

  // ---- QKV projections (V written transposed for attention) ----
  dim3 gD(D / 64, M / 128);
  gemm_wmma<true, false, false><<<gD, 256, 0, stream>>>(xb, Wqt, bq, Qb, M, D, D);
  gemm_wmma<true, false, false><<<gD, 256, 0, stream>>>(xb, Wkt, bk, Kb, M, D, D);
  gemm_wmma<true, false, true ><<<gD, 256, 0, stream>>>(xb, Wvt, bv, Vtb, M, D, D);

  // ---- flash attention ----
  dim3 gA(T / 128, N_HEADS, B);
  attn_kernel<<<gA, 256, 0, stream>>>(Qb, Kb, Vtb, mask, Ab, T, M);

  // ---- output projection + residual + LN1 ----
  gemm_wmma<false, false, false><<<gD, 256, 0, stream>>>(Ab, Wot, bo, tmpf, M, D, D);
  ln_kernel<<<M, 256, 0, stream>>>(x, tmpf, ln1_g, ln1_b, x1, x1b);

  // ---- FFN ----
  dim3 gH(H / 64, M / 128);
  gemm_wmma<true, true, false><<<gH, 256, 0, stream>>>(x1b, W1t, b1, hb, M, H, D);
  gemm_wmma<false, false, false><<<gD, 256, 0, stream>>>(hb, W2t, b2, tmpf, M, D, H);

  // ---- residual + LN2 -> final f32 output ----
  ln_kernel<<<M, 256, 0, stream>>>(x1, tmpf, ln2_g, ln2_b, (float*)d_out,
                                   (__bf16*)nullptr);
}
